// LSTMAttention_19902878450127
// MI455X (gfx1250) — matrix operand order repaired and checked by reading, not compiled, not run on hardware
//
#include <hip/hip_runtime.h>
#include <hip/hip_bf16.h>
#include <math.h>

typedef __attribute__((ext_vector_type(16))) __bf16 v16bf;
typedef __attribute__((ext_vector_type(8)))  float  v8f;

// async-to-LDS builtin operand types (probe-confirmed: param is pointer to
// 4 x i32 vector; AS1 source, AS3 destination)
typedef int v4i_ __attribute__((vector_size(16)));
typedef __attribute__((address_space(1))) v4i_ gv4i;
typedef __attribute__((address_space(3))) v4i_ lv4i;

#define B_DIM 32
#define H_DIM 512
#define S_DIM 2048
#define F_DIM 1024
#define A_DIM 512

#define MROWS 32            // rows per block (2 x 16 M-tiles per wave)
#define KCHUNK 256          // K elements staged per LDS chunk
#define TSTRIDE (KCHUNK+8)  // +8 floats pad (bank spread)

#if __has_builtin(__builtin_amdgcn_global_load_async_to_lds_b128)
#define ASYNC_LDS 1
#else
#define ASYNC_LDS 0
#endif

// ---------------- kernel 0: proj_h[b,a] = h[b,:] @ W2_w + W2_b ----------------
__global__ __launch_bounds__(256) void k_proj_h(const float* __restrict__ hidden,
                                                const float* __restrict__ W2w,
                                                const float* __restrict__ W2b,
                                                float* __restrict__ projh) {
  int idx = blockIdx.x * 256 + threadIdx.x;      // 16384 total
  int b = idx >> 9, a = idx & 511;
  float sum = W2b[a];
  #pragma unroll
  for (int d = 0; d < 2; ++d) {
    const float* hrow = hidden + ((size_t)d * B_DIM + b) * H_DIM;   // hidden[d,b,:]
    const float* wcol = W2w + (size_t)d * H_DIM * A_DIM + a;        // W2_w[d*512+j, a]
    for (int j = 0; j < H_DIM; ++j)
      sum += hrow[j] * wcol[(size_t)j * A_DIM];
  }
  projh[idx] = sum;
}

// ------------- kernel 0b: W1t[a,k] = bf16(W1_w[k,a])  (K-major per column) ----
__global__ __launch_bounds__(256) void k_w1t(const float* __restrict__ W1w,
                                             unsigned short* __restrict__ W1t_raw) {
  __bf16* W1t = (__bf16*)W1t_raw;
  int idx = blockIdx.x * 256 + threadIdx.x;      // 524288 total
  int a = idx >> 10, k = idx & 1023;
  W1t[idx] = (__bf16)W1w[(size_t)k * A_DIM + a];
}

// ------------- kernel 1: fused GEMM + bias + tanh + V-dot -> score[B*S] -------
// Block: 128 threads (4 waves). Block tile: 32 rows x 512 cols.
// Wave: 2 M-tiles (32 rows) x 8 N-tiles (128 cols) -> each B fragment feeds
// 2 WMMAs (register blocking halves B traffic vs reuse-1).
__global__ __launch_bounds__(128) void k_score(const float* __restrict__ enc,
                                               const unsigned short* __restrict__ W1t_raw,
                                               const float* __restrict__ W1b,
                                               const float* __restrict__ projh,
                                               const float* __restrict__ Vw,
                                               const float* __restrict__ Vb,
                                               float* __restrict__ score) {
  const __bf16* W1t = (const __bf16*)W1t_raw;
  __shared__ float tile[MROWS * TSTRIDE] __attribute__((aligned(16)));
  __shared__ float red[4][MROWS];

  const int t    = threadIdx.x;
  const int lane = t & 31;
  const int wv   = t >> 5;        // wave 0..3 -> columns [wv*128, wv*128+128)
  const int nlo  = lane & 15;     // A-matrix row within tile / B-matrix column
  const int g    = lane >> 4;     // K half-group
  const int row0 = blockIdx.x * MROWS;
  const int b    = row0 >> 11;    // row0 / S_DIM

  v8f acc[2][8] = {};

  for (int chunk = 0; chunk < 4; ++chunk) {
    if (chunk) __syncthreads();
    // stage 32 rows x 256 f32 of enc_out into LDS (async on gfx1250)
    #pragma unroll
    for (int i = 0; i < 16; ++i) {
      int c    = t + 128 * i;     // 2048 x 16B transfers
      int row  = c >> 6;          // 64 float4 per row
      int col4 = c & 63;
      const float* src = enc + (size_t)(row0 + row) * F_DIM + chunk * KCHUNK + col4 * 4;
      float*       dst = &tile[row * TSTRIDE + col4 * 4];
#if ASYNC_LDS
      __builtin_amdgcn_global_load_async_to_lds_b128((gv4i*)src, (lv4i*)dst, 0, 0);
#else
      *(float4*)dst = *(const float4*)src;
#endif
    }
#if ASYNC_LDS
#if __has_builtin(__builtin_amdgcn_s_wait_asynccnt)
    __builtin_amdgcn_s_wait_asynccnt(0);
#else
    asm volatile("s_wait_asynccnt 0" ::: "memory");
#endif
#endif
    __syncthreads();

    for (int kt = 0; kt < KCHUNK / 32; ++kt) {
      const int kb = kt * 32;
      // A fragments, 16x32 bf16 each: lane holds row m = mt*16+nlo,
      // K = {16g..16g+15}
      v16bf a[2];
      #pragma unroll
      for (int mt = 0; mt < 2; ++mt) {
        const float* rp = &tile[(mt * 16 + nlo) * TSTRIDE + kb + 16 * g];
        float4 f0 = *(const float4*)(rp);
        float4 f1 = *(const float4*)(rp + 4);
        float4 f2 = *(const float4*)(rp + 8);
        float4 f3 = *(const float4*)(rp + 12);
        a[mt][0]  = (__bf16)f0.x; a[mt][1]  = (__bf16)f0.y; a[mt][2]  = (__bf16)f0.z; a[mt][3]  = (__bf16)f0.w;
        a[mt][4]  = (__bf16)f1.x; a[mt][5]  = (__bf16)f1.y; a[mt][6]  = (__bf16)f1.z; a[mt][7]  = (__bf16)f1.w;
        a[mt][8]  = (__bf16)f2.x; a[mt][9]  = (__bf16)f2.y; a[mt][10] = (__bf16)f2.z; a[mt][11] = (__bf16)f2.w;
        a[mt][12] = (__bf16)f3.x; a[mt][13] = (__bf16)f3.y; a[mt][14] = (__bf16)f3.z; a[mt][15] = (__bf16)f3.w;
      }

      const int kglob = chunk * KCHUNK + kb + 16 * g;   // lane's contiguous 16-K run
      #pragma unroll
      for (int nt = 0; nt < 8; ++nt) {
        int col = wv * 128 + nt * 16 + nlo;             // B-matrix column N
        const v16bf bm = *(const v16bf*)(W1t + (size_t)col * F_DIM + kglob);
        acc[0][nt] = __builtin_amdgcn_wmma_f32_16x16x32_bf16(
            false, a[0], false, bm, (short)0, acc[0][nt], false, false);
        acc[1][nt] = __builtin_amdgcn_wmma_f32_16x16x32_bf16(
            false, a[1], false, bm, (short)0, acc[1][nt], false, false);
      }
    }
  }

  // fused epilogue: tanh(proj + W1_b + proj_h) . V_w  reduced over A
  float s[2][8] = {};
  #pragma unroll
  for (int nt = 0; nt < 8; ++nt) {
    int col = wv * 128 + nt * 16 + nlo;
    float bias = W1b[col] + projh[b * A_DIM + col];
    float vw   = Vw[col];
    #pragma unroll
    for (int mt = 0; mt < 2; ++mt) {
      #pragma unroll
      for (int r = 0; r < 8; ++r) {                    // row m = mt*16 + r + 8*g
        float e = tanhf(acc[mt][nt][r] + bias);
        s[mt][r] += e * vw;
      }
    }
  }
  // butterfly over the 16 lanes of each half (N dimension)
  #pragma unroll
  for (int mask = 1; mask <= 8; mask <<= 1) {
    #pragma unroll
    for (int mt = 0; mt < 2; ++mt)
      #pragma unroll
      for (int r = 0; r < 8; ++r) s[mt][r] += __shfl_xor(s[mt][r], mask, 32);
  }
  if (nlo == 0) {
    #pragma unroll
    for (int mt = 0; mt < 2; ++mt)
      #pragma unroll
      for (int r = 0; r < 8; ++r) red[wv][mt * 16 + 8 * g + r] = s[mt][r];
  }
  __syncthreads();
  if (t < MROWS) {
    float sum = red[0][t] + red[1][t] + red[2][t] + red[3][t] + Vb[0];
    score[row0 + t] = sum;
  }
}

// ---------------- kernel 2a: softmax over S per batch ----------------
__global__ __launch_bounds__(256) void k_softmax(const float* __restrict__ score,
                                                 float* __restrict__ attn) {
  __shared__ float red[256];
  int b = blockIdx.x, t = threadIdx.x;
  float v[8];
  float m = -INFINITY;
  #pragma unroll
  for (int i = 0; i < 8; ++i) { v[i] = score[b * S_DIM + i * 256 + t]; m = fmaxf(m, v[i]); }
  red[t] = m; __syncthreads();
  for (int o = 128; o > 0; o >>= 1) { if (t < o) red[t] = fmaxf(red[t], red[t + o]); __syncthreads(); }
  float M = red[0]; __syncthreads();
  float s = 0.f;
  #pragma unroll
  for (int i = 0; i < 8; ++i) { v[i] = __expf(v[i] - M); s += v[i]; }
  red[t] = s; __syncthreads();
  for (int o = 128; o > 0; o >>= 1) { if (t < o) red[t] += red[t + o]; __syncthreads(); }
  float inv = 1.0f / red[0];
  #pragma unroll
  for (int i = 0; i < 8; ++i) attn[b * S_DIM + i * 256 + t] = v[i] * inv;
}

// ---------------- kernel 2b: partial context over S chunks (deterministic) ----
__global__ __launch_bounds__(256) void k_ctx_partial(const float* __restrict__ attn,
                                                     const float* __restrict__ enc,
                                                     float* __restrict__ part) {
  __shared__ float aw[128];
  int b = blockIdx.x >> 4, ch = blockIdx.x & 15, t = threadIdx.x;
  if (t < 128) aw[t] = attn[b * S_DIM + ch * 128 + t];
  __syncthreads();
  float4 acc = {0.f, 0.f, 0.f, 0.f};
  int f0 = t * 4;
  const float* base = enc + ((size_t)b * S_DIM + ch * 128) * F_DIM + f0;
  for (int s = 0; s < 128; ++s) {
    float w = aw[s];
    float4 v = *(const float4*)(base + (size_t)s * F_DIM);
    acc.x += w * v.x; acc.y += w * v.y; acc.z += w * v.z; acc.w += w * v.w;
  }
  *(float4*)(part + (size_t)blockIdx.x * F_DIM + f0) = acc;
}

// ---------------- kernel 2c: reduce 16 chunks -> context[B,F] ----------------
__global__ __launch_bounds__(256) void k_ctx_reduce(const float* __restrict__ part,
                                                    float* __restrict__ out) {
  int idx = blockIdx.x * 256 + threadIdx.x;   // 32768 total
  int b = idx >> 10, f = idx & 1023;
  float s = 0.f;
  #pragma unroll
  for (int c = 0; c < 16; ++c) s += part[((size_t)(b * 16 + c)) * F_DIM + f];
  out[idx] = s;
}

extern "C" void kernel_launch(void* const* d_in, const int* in_sizes, int n_in,
                              void* d_out, int out_size, void* d_ws, size_t ws_size,
                              hipStream_t stream) {
  const float* hidden = (const float*)d_in[0];
  const float* enc    = (const float*)d_in[1];
  const float* W1w    = (const float*)d_in[2];
  const float* W1b    = (const float*)d_in[3];
  const float* W2w    = (const float*)d_in[4];
  const float* W2b    = (const float*)d_in[5];
  const float* Vw     = (const float*)d_in[6];
  const float* Vb     = (const float*)d_in[7];
  float* out = (float*)d_out;

  char* ws = (char*)d_ws;
  float*          projh = (float*)(ws);                                   //  64 KB
  unsigned short* W1t   = (unsigned short*)(ws + 65536);                  //   1 MB
  float*          score = (float*)(ws + 65536 + 1048576);                 // 256 KB
  float*          attn  = (float*)(ws + 65536 + 1048576 + 262144);        // 256 KB
  float*          part  = (float*)(ws + 65536 + 1048576 + 2 * 262144);    //   2 MB

  k_proj_h     <<<64,   256, 0, stream>>>(hidden, W2w, W2b, projh);
  k_w1t        <<<2048, 256, 0, stream>>>(W1w, W1t);
  k_score      <<<2048, 128, 0, stream>>>(enc, W1t, W1b, projh, Vw, Vb, score);
  k_softmax    <<<32,   256, 0, stream>>>(score, attn);
  k_ctx_partial<<<512,  256, 0, stream>>>(attn, enc, part);
  k_ctx_reduce <<<128,  256, 0, stream>>>(part, out);
}
